// DepthTransformerLayer_16896401342953
// MI455X (gfx1250) — matrix operand order repaired and checked by reading, not compiled, not run on hardware
//
#include <hip/hip_runtime.h>
#include <stdint.h>

typedef unsigned short u16;
typedef __attribute__((ext_vector_type(16))) __bf16 v16bf;
typedef __attribute__((ext_vector_type(8)))  float  v8f;
typedef int i4v __attribute__((vector_size(16)));   // matches builtin param type

enum { Bsz = 8192, Kdep = 8, Dm = 512, Mlp = 2048, Nh = 8, Hd = 64 };

// Async global->LDS path (gfx1250): gated so a missing builtin degrades to a
// register-pipelined copy instead of failing the build.
#if defined(__AMDGCN__) && \
    __has_builtin(__builtin_amdgcn_global_load_async_to_lds_b128) && \
    __has_builtin(__builtin_amdgcn_s_wait_asynccnt)
#define USE_ASYNC 1
#else
#define USE_ASYNC 0
#endif

#if USE_ASYNC
#define ASYNC_GPTR(p) ((__attribute__((address_space(1))) i4v*)(p))
#define ASYNC_LPTR(p) ((__attribute__((address_space(3))) i4v*)(p))
#endif

__device__ __forceinline__ u16 f2bf(float f) {
  uint32_t u = __builtin_bit_cast(uint32_t, f);
  u += 0x7FFFu + ((u >> 16) & 1u);           // round-to-nearest-even
  return (u16)(u >> 16);
}
__device__ __forceinline__ float bf2f(u16 h) {
  uint32_t u = ((uint32_t)h) << 16;
  return __builtin_bit_cast(float, u);
}
__device__ __forceinline__ void unpack8(uint4 u, float* f) {
  f[0] = bf2f((u16)u.x); f[1] = bf2f((u16)(u.x >> 16));
  f[2] = bf2f((u16)u.y); f[3] = bf2f((u16)(u.y >> 16));
  f[4] = bf2f((u16)u.z); f[5] = bf2f((u16)(u.z >> 16));
  f[6] = bf2f((u16)u.w); f[7] = bf2f((u16)(u.w >> 16));
}

// ---------------------------------------------------------------------------
// rmsnorm over rows of length 512: one wave32 per row (16 elems / lane).
// g is folded into the weights elsewhere, so this writes xhat = x*rsqrt(ms).
// ---------------------------------------------------------------------------
__global__ __launch_bounds__(256) void rmsnorm_rows(
    const float* __restrict__ x, u16* __restrict__ out, int rows,
    const int* __restrict__ ad, int layer) {
  if (layer >= *ad) return;
  int gtid = blockIdx.x * 256 + threadIdx.x;
  int row  = gtid >> 5, lane = gtid & 31;
  if (row >= rows) return;
  const float* xr = x + (size_t)row * Dm;
  float4 v[4];
  float ss = 0.f;
#pragma unroll
  for (int c = 0; c < 4; ++c) {
    v[c] = *(const float4*)(xr + c * 128 + lane * 4);
    ss += v[c].x * v[c].x + v[c].y * v[c].y + v[c].z * v[c].z + v[c].w * v[c].w;
  }
#pragma unroll
  for (int m = 16; m >= 1; m >>= 1) ss += __shfl_xor(ss, m, 32);
  float sc = rsqrtf(ss * (1.0f / Dm) + 1e-6f);
  u16* orow = out + (size_t)row * Dm;
#pragma unroll
  for (int c = 0; c < 4; ++c) {
    *(ushort4*)(orow + c * 128 + lane * 4) =
        make_ushort4(f2bf(v[c].x * sc), f2bf(v[c].y * sc),
                     f2bf(v[c].z * sc), f2bf(v[c].w * sc));
  }
}

// ---------------------------------------------------------------------------
// Convert f32 weight W (rowsK x colsN, row-major) to bf16 TRANSPOSED layout
// outT (colsN x rowsK, K contiguous), scaling W row k by g[k] (rmsnorm fold).
// K-contiguous B rows make the GEMM's B-staging a pure b128 copy (async-able).
// ---------------------------------------------------------------------------
__global__ __launch_bounds__(256) void convert_wT(
    const float* __restrict__ W, const float* __restrict__ g,
    u16* __restrict__ outT, int rowsK, int colsN, int total4,
    const int* __restrict__ ad, int layer) {
  if (layer >= *ad) return;
  int t = blockIdx.x * 256 + threadIdx.x;
  if (t >= total4) return;
  int n  = t % colsN;              // coalesced reads across adjacent threads
  int k0 = (t / colsN) * 4;
  u16 o[4];
#pragma unroll
  for (int i = 0; i < 4; ++i) {
    float s = g ? g[k0 + i] : 1.0f;
    o[i] = f2bf(W[(size_t)(k0 + i) * colsN + n] * s);
  }
  *(ushort4*)(outT + (size_t)n * rowsK + k0) = make_ushort4(o[0], o[1], o[2], o[3]);
}

// ---------------------------------------------------------------------------
// Stage one 128x32 bf16 tile of A (row-major, lda) and one of B^T (row-major
// N x K, ldb) into LDS. 512 chunks of 16B each per operand; 2 per thread.
// ---------------------------------------------------------------------------
__device__ __forceinline__ void stage_tile(
    u16 (*Asb)[40], u16 (*Bsb)[40],
    const u16* __restrict__ A, int lda,
    const u16* __restrict__ Bt, int ldb,
    size_t tileM, int tileN, int kt, int tid) {
#if USE_ASYNC
#pragma unroll
  for (int it = 0; it < 2; ++it) {
    int c = tid + it * 256;
    int r = c >> 2, c0 = (c & 3) << 3;
    const u16* ga = A  + (tileM + r) * (size_t)lda + kt + c0;
    const u16* gb = Bt + (size_t)(tileN + r) * ldb + kt + c0;
    __builtin_amdgcn_global_load_async_to_lds_b128(
        ASYNC_GPTR(ga), ASYNC_LPTR(&Asb[r][c0]), 0, 0);
    __builtin_amdgcn_global_load_async_to_lds_b128(
        ASYNC_GPTR(gb), ASYNC_LPTR(&Bsb[r][c0]), 0, 0);
  }
#else
  uint4 ra[2], rb[2];
#pragma unroll
  for (int it = 0; it < 2; ++it) {           // issue all loads back-to-back
    int c = tid + it * 256;
    int r = c >> 2, c0 = (c & 3) << 3;
    ra[it] = *(const uint4*)(A  + (tileM + r) * (size_t)lda + kt + c0);
    rb[it] = *(const uint4*)(Bt + (size_t)(tileN + r) * ldb + kt + c0);
  }
#pragma unroll
  for (int it = 0; it < 2; ++it) {
    int c = tid + it * 256;
    int r = c >> 2, c0 = (c & 3) << 3;
    *(uint4*)&Asb[r][c0] = ra[it];
    *(uint4*)&Bsb[r][c0] = rb[it];
  }
#endif
}

// ---------------------------------------------------------------------------
// bf16 GEMM  C[M,N] = A[M,K] * B[K,N] with B supplied TRANSPOSED (N x K).
// 256 threads = 8 wave32s; block tile 128x128, K-step 32, LDS double-buffered
// (async global->LDS when available). Wave grid 4(M) x 2(N); each wave does
// 2x4 v_wmma_f32_16x16x32_bf16 per K-step.
// EPI: 0 = store bf16, 2 = store f32 + f32 residual add.
// ---------------------------------------------------------------------------
template <int EPI>
__global__ __launch_bounds__(256) void gemm_bf16(
    const u16* __restrict__ A, int lda,
    const u16* __restrict__ Bt, int ldb,
    void* __restrict__ Cv, int ldc,
    const float* __restrict__ Res, int ldres,
    int Kd, const int* __restrict__ ad, int layer) {
  if (layer >= *ad) return;

  __shared__ u16 As[2][128][40];   // 10KB per buffer, 16B-aligned rows
  __shared__ u16 Bs[2][128][40];

  const int tid   = threadIdx.x;
  const int lane  = tid & 31;
  const int waveM = (tid >> 5) & 3;
  const int waveN = (tid >> 5) >> 2;
  const size_t tileM = (size_t)blockIdx.y * 128;
  const int    tileN = blockIdx.x * 128;

  v8f acc[2][4] = {};

  const int nK = Kd >> 5;
  int buf = 0;
  stage_tile(As[0], Bs[0], A, lda, Bt, ldb, tileM, tileN, 0, tid);

  for (int t = 0; t < nK; ++t) {
#if USE_ASYNC
    __builtin_amdgcn_s_wait_asynccnt(0);
#endif
    __syncthreads();   // staged buffer visible to all waves; prev reads done
    if (t + 1 < nK)
      stage_tile(As[buf ^ 1], Bs[buf ^ 1], A, lda, Bt, ldb,
                 tileM, tileN, (t + 1) << 5, tid);

    const u16 (*Asb)[40] = As[buf];
    const u16 (*Bsb)[40] = Bs[buf];
    union Frag { uint4 u[2]; v16bf v; };
    Frag af[2], bfr[4];
    const int hsel  = lane >> 4;
    const int arow0 = waveM * 32 + (lane & 15);
    const int ac0   = hsel * 8;            // A: K cols 0..7 / 8..15 (+16 pair)
#pragma unroll
    for (int fm = 0; fm < 2; ++fm) {
      int r = arow0 + fm * 16;
      af[fm].u[0] = *(const uint4*)&Asb[r][ac0];
      af[fm].u[1] = *(const uint4*)&Asb[r][ac0 + 16];
    }
    const int bn0 = waveN * 64 + (lane & 15);
    const int bk0 = hsel * 8;              // B: K rows 0..7 / 8..15 (+16 pair)
#pragma unroll
    for (int fn = 0; fn < 4; ++fn) {
      int n = bn0 + fn * 16;
      bfr[fn].u[0] = *(const uint4*)&Bsb[n][bk0];
      bfr[fn].u[1] = *(const uint4*)&Bsb[n][bk0 + 16];
    }
#pragma unroll
    for (int fm = 0; fm < 2; ++fm)
#pragma unroll
      for (int fn = 0; fn < 4; ++fn)
        acc[fm][fn] = __builtin_amdgcn_wmma_f32_16x16x32_bf16(
            false, af[fm].v, false, bfr[fn].v,
            (short)0, acc[fm][fn], false, false);
    buf ^= 1;
  }

  // writeback: lane holds col = lane&15, rows (lane>>4)*8 + i
  const int col    = lane & 15;
  const int rowoff = (lane >> 4) * 8;
#pragma unroll
  for (int fm = 0; fm < 2; ++fm) {
#pragma unroll
    for (int fn = 0; fn < 4; ++fn) {
      size_t gm0 = tileM + waveM * 32 + fm * 16 + rowoff;
      int    gn  = tileN + waveN * 64 + fn * 16 + col;
#pragma unroll
      for (int i = 0; i < 8; ++i) {
        size_t m = gm0 + i;
        float val = acc[fm][fn][i];
        if (EPI == 0) {
          ((u16*)Cv)[m * (size_t)ldc + gn] = f2bf(val);
        } else {
          ((float*)Cv)[m * (size_t)ldc + gn] =
              val + Res[m * (size_t)ldres + gn];
        }
      }
    }
  }
}

// ---------------------------------------------------------------------------
// Tiny attention: last-position query vs L<=8 keys, per (batch, head) thread.
// ---------------------------------------------------------------------------
__global__ __launch_bounds__(256) void attn_small(
    const u16* __restrict__ q, const u16* __restrict__ kb,
    const u16* __restrict__ vb, u16* __restrict__ ob, int L,
    const int* __restrict__ ad, int layer) {
  if (layer >= *ad) return;
  int t = blockIdx.x * 256 + threadIdx.x;   // 0 .. B*H-1
  int b = t >> 3, h = t & 7;
  const u16* qr = q + (size_t)b * Dm + h * Hd;
  float qv[64];
#pragma unroll
  for (int c = 0; c < 8; ++c) unpack8(*(const uint4*)(qr + c * 8), qv + c * 8);

  float s[Kdep];
  float mx = -3.0e38f;
  for (int j = 0; j < L; ++j) {
    const u16* kr = kb + (size_t)b * (Kdep * Dm) + (size_t)j * Dm + h * Hd;
    float accd = 0.f;
#pragma unroll
    for (int c = 0; c < 8; ++c) {
      float kv[8];
      unpack8(*(const uint4*)(kr + c * 8), kv);
#pragma unroll
      for (int i = 0; i < 8; ++i) accd += qv[c * 8 + i] * kv[i];
    }
    s[j] = accd * 0.125f;                   // HD^-0.5
    mx = fmaxf(mx, s[j]);
  }
  float se = 0.f;
  for (int j = 0; j < L; ++j) { s[j] = __expf(s[j] - mx); se += s[j]; }
  float inv = 1.0f / se;

  float o[64];
#pragma unroll
  for (int d = 0; d < 64; ++d) o[d] = 0.f;
  for (int j = 0; j < L; ++j) {
    float p = s[j] * inv;
    const u16* vr = vb + (size_t)b * (Kdep * Dm) + (size_t)j * Dm + h * Hd;
#pragma unroll
    for (int c = 0; c < 8; ++c) {
      float vv[8];
      unpack8(*(const uint4*)(vr + c * 8), vv);
#pragma unroll
      for (int i = 0; i < 8; ++i) o[c * 8 + i] += p * vv[i];
    }
  }
  u16* orow = ob + (size_t)b * Dm + h * Hd;
#pragma unroll
  for (int c = 0; c < 16; ++c)
    *(ushort4*)(orow + c * 4) = make_ushort4(
        f2bf(o[c * 4 + 0]), f2bf(o[c * 4 + 1]),
        f2bf(o[c * 4 + 2]), f2bf(o[c * 4 + 3]));
}

// h = silu(g) * u, bf16 in/out
__global__ __launch_bounds__(256) void silu_mul(
    const u16* __restrict__ g, const u16* __restrict__ u,
    u16* __restrict__ h, const int* __restrict__ ad, int layer) {
  if (layer >= *ad) return;
  size_t i = ((size_t)blockIdx.x * 256 + threadIdx.x) * 4;
  ushort4 gv = *(const ushort4*)(g + i);
  ushort4 uv = *(const ushort4*)(u + i);
  float g0 = bf2f(gv.x), g1 = bf2f(gv.y), g2 = bf2f(gv.z), g3 = bf2f(gv.w);
  float r0 = g0 / (1.f + __expf(-g0)) * bf2f(uv.x);
  float r1 = g1 / (1.f + __expf(-g1)) * bf2f(uv.y);
  float r2 = g2 / (1.f + __expf(-g2)) * bf2f(uv.z);
  float r3 = g3 / (1.f + __expf(-g3)) * bf2f(uv.w);
  *(ushort4*)(h + i) = make_ushort4(f2bf(r0), f2bf(r1), f2bf(r2), f2bf(r3));
}

// ---------------------------------------------------------------------------
extern "C" void kernel_launch(void* const* d_in, const int* in_sizes, int n_in,
                              void* d_out, int out_size, void* d_ws,
                              size_t ws_size, hipStream_t stream) {
  (void)in_sizes; (void)n_in; (void)out_size; (void)ws_size;
  const float* x      = (const float*)d_in[0];
  const float* Wq     = (const float*)d_in[2];
  const float* Wk     = (const float*)d_in[3];
  const float* Wv     = (const float*)d_in[4];
  const float* Wo     = (const float*)d_in[5];
  const float* Wg     = (const float*)d_in[6];
  const float* Wu     = (const float*)d_in[7];
  const float* Wd     = (const float*)d_in[8];
  const float* g_attn = (const float*)d_in[9];
  const float* g_ffn  = (const float*)d_in[10];
  const int*   ad     = (const int*)d_in[11];
  float* out = (float*)d_out;

  char* ws = (char*)d_ws;
  size_t off = 0;
  auto alloc = [&](size_t bytes) -> char* {
    char* p = ws + off;
    off += (bytes + 255) & ~(size_t)255;
    return p;
  };
  u16*   xhat  = (u16*)alloc((size_t)Bsz * Kdep * Dm * 2);  // 64 MB
  u16*   wq    = (u16*)alloc((size_t)Dm * Dm * 2);          // all W buffers
  u16*   wk    = (u16*)alloc((size_t)Dm * Dm * 2);          // hold W^T bf16
  u16*   wv    = (u16*)alloc((size_t)Dm * Dm * 2);
  u16*   wo    = (u16*)alloc((size_t)Dm * Dm * 2);
  u16*   wg    = (u16*)alloc((size_t)Dm * Mlp * 2);
  u16*   wu    = (u16*)alloc((size_t)Dm * Mlp * 2);
  u16*   wd    = (u16*)alloc((size_t)Mlp * Dm * 2);
  u16*   qbuf  = (u16*)alloc((size_t)Bsz * Dm * 2);
  u16*   kbuf  = (u16*)alloc((size_t)Bsz * Kdep * Dm * 2);  // 64 MB
  u16*   vbuf  = (u16*)alloc((size_t)Bsz * Kdep * Dm * 2);  // 64 MB
  u16*   obuf  = (u16*)alloc((size_t)Bsz * Dm * 2);
  float* xmid  = (float*)alloc((size_t)Bsz * Dm * 4);
  u16*   xhat2 = (u16*)alloc((size_t)Bsz * Dm * 2);
  u16*   gbuf  = (u16*)alloc((size_t)Bsz * Mlp * 2);
  u16*   ubuf  = (u16*)alloc((size_t)Bsz * Mlp * 2);
  u16*   hbuf  = (u16*)alloc((size_t)Bsz * Mlp * 2);

  const dim3 blk(256);
  const dim3 gridD(Dm / 128, Bsz / 128);     // N=512 GEMMs
  const dim3 gridM(Mlp / 128, Bsz / 128);    // N=2048 GEMMs
  const int wAttn4 = Dm * Dm / 4;            // 65536
  const int wMlp4  = Dm * Mlp / 4;           // 262144

  // xhat for all (b, j) rows — k-independent part of rmsnorm
  rmsnorm_rows<<<(Bsz * Kdep) / 8, blk, 0, stream>>>(x, xhat, Bsz * Kdep, ad, 0);

  for (int k = 0; k < Kdep; ++k) {
    const int L = k + 1;
    const size_t wOffD = (size_t)k * Dm * Dm;
    const size_t wOffM = (size_t)k * Dm * Mlp;
    // weights -> bf16 W^T, folding g_attn/g_ffn into original rows (K-dim)
    convert_wT<<<wAttn4 / 256, blk, 0, stream>>>(Wq + wOffD, g_attn + k * Dm, wq, Dm,  Dm,  wAttn4, ad, k);
    convert_wT<<<wAttn4 / 256, blk, 0, stream>>>(Wk + wOffD, g_attn + k * Dm, wk, Dm,  Dm,  wAttn4, ad, k);
    convert_wT<<<wAttn4 / 256, blk, 0, stream>>>(Wv + wOffD, g_attn + k * Dm, wv, Dm,  Dm,  wAttn4, ad, k);
    convert_wT<<<wAttn4 / 256, blk, 0, stream>>>(Wo + wOffD, nullptr,         wo, Dm,  Dm,  wAttn4, ad, k);
    convert_wT<<<wMlp4 / 256,  blk, 0, stream>>>(Wg + wOffM, g_ffn + k * Dm,  wg, Dm,  Mlp, wMlp4,  ad, k);
    convert_wT<<<wMlp4 / 256,  blk, 0, stream>>>(Wu + wOffM, g_ffn + k * Dm,  wu, Dm,  Mlp, wMlp4,  ad, k);
    convert_wT<<<wMlp4 / 256,  blk, 0, stream>>>(Wd + wOffM, nullptr,         wd, Mlp, Dm,  wMlp4,  ad, k);

    // Q at position k only  (B^T has ldb = Kdim)
    gemm_bf16<0><<<gridD, blk, 0, stream>>>(xhat + k * Dm, Kdep * Dm, wq, Dm,
                                            qbuf, Dm, nullptr, 0, Dm, ad, k);
    // K,V for positions 0..k
    for (int j = 0; j < L; ++j) {
      gemm_bf16<0><<<gridD, blk, 0, stream>>>(xhat + j * Dm, Kdep * Dm, wk, Dm,
                                              kbuf + j * Dm, Kdep * Dm, nullptr, 0, Dm, ad, k);
      gemm_bf16<0><<<gridD, blk, 0, stream>>>(xhat + j * Dm, Kdep * Dm, wv, Dm,
                                              vbuf + j * Dm, Kdep * Dm, nullptr, 0, Dm, ad, k);
    }
    attn_small<<<(Bsz * Nh) / 256, blk, 0, stream>>>(qbuf, kbuf, vbuf, obuf, L, ad, k);

    // xmid = x[:,k,:] + o @ Wo
    gemm_bf16<2><<<gridD, blk, 0, stream>>>(obuf, Dm, wo, Dm,
                                            xmid, Dm, x + k * Dm, Kdep * Dm, Dm, ad, k);
    rmsnorm_rows<<<Bsz / 8, blk, 0, stream>>>(xmid, xhat2, Bsz, ad, k);

    gemm_bf16<0><<<gridM, blk, 0, stream>>>(xhat2, Dm, wg, Dm,
                                            gbuf, Mlp, nullptr, 0, Dm, ad, k);
    gemm_bf16<0><<<gridM, blk, 0, stream>>>(xhat2, Dm, wu, Dm,
                                            ubuf, Mlp, nullptr, 0, Dm, ad, k);
    silu_mul<<<(Bsz * Mlp / 4) / 256, blk, 0, stream>>>(gbuf, ubuf, hbuf, ad, k);

    // out[:,k,:] = xmid + h @ Wd
    gemm_bf16<2><<<gridD, blk, 0, stream>>>(hbuf, Mlp, wd, Mlp,
                                            out + k * Dm, Kdep * Dm, xmid, Dm,
                                            Mlp, ad, k);
  }
}
